// SpectralDense_58188216926786
// MI455X (gfx1250) — compile-verified, matrix operand order checked by reading, not run on hardware
//
#include <hip/hip_runtime.h>
#include <hip/hip_bf16.h>

// ---------------------------------------------------------------------------
// SpectralDense on MI455X (gfx1250, wave32, WMMA)
//   sigma = top singular value of K (power iteration on K^T K)
//   out   = inputs @ (K / sigma)
// Main GEMM: M=8192, N=2048, K=2048 -> 68.7 GFLOP, ~151 MB traffic
// (~455 FLOP/B): near the f16-WMMA/HBM roofline crossover, so
// v_wmma_f32_16x16x32_f16 with fp32 accumulate is the right precision path.
// Double-buffered LDS + register staging overlaps global loads with WMMA.
// ---------------------------------------------------------------------------

typedef __attribute__((ext_vector_type(16))) _Float16 v16h;
typedef __attribute__((ext_vector_type(8)))  _Float16 v8h;
typedef __attribute__((ext_vector_type(4)))  _Float16 v4h;
typedef __attribute__((ext_vector_type(8)))  float    v8f;

#define D_IN   2048
#define FEATS  2048
#define M_TOT  8192          // BATCH * SEQ = 4 * 2048
#define BM     128
#define BN     128
#define BK     32
#define LPITCH 40            // LDS row pitch in halves: 80B (16B-aligned, 20-bank stride)
#define PI_ITERS 64

// ------------------------------- pre-pass ----------------------------------
// Bt[n][k] = (f16) K[k][n]   (32x32 tiled transpose via LDS)
__global__ __launch_bounds__(256) void sd_transpose_cvt(
    const float* __restrict__ Kmat, _Float16* __restrict__ Bt) {
  __shared__ float tile[32][33];
  const int tx = threadIdx.x & 31;
  const int ty = threadIdx.x >> 5;              // 0..7
  const int kb = blockIdx.y * 32;               // k base
  const int nb = blockIdx.x * 32;               // n base
#pragma unroll
  for (int j = 0; j < 4; ++j)
    tile[ty + j * 8][tx] = Kmat[(size_t)(kb + ty + j * 8) * FEATS + nb + tx];
  __syncthreads();
#pragma unroll
  for (int j = 0; j < 4; ++j)
    Bt[(size_t)(nb + ty + j * 8) * D_IN + kb + tx] =
        (_Float16)tile[tx][ty + j * 8];
}

// --------------------------- power iteration -------------------------------
__global__ void sd_init_v(float* __restrict__ v) {
  int j = blockIdx.x * 256 + threadIdx.x;
  unsigned h = (unsigned)j * 2654435761u;
  h ^= h >> 16; h *= 2246822519u; h ^= h >> 13;
  v[j] = (float)(h & 0xFFFF) * (1.0f / 65536.0f) - 0.5f;   // deterministic +-
}

// w[i] = sum_j K[i][j] * v[j]      grid = 2048 blocks x 256
__global__ __launch_bounds__(256) void sd_mv_Kv(
    const float* __restrict__ Kmat, const float* __restrict__ v,
    float* __restrict__ w) {
  __shared__ float red[256];
  const int i = blockIdx.x;
  float s = 0.f;
  for (int j = threadIdx.x; j < FEATS; j += 256)
    s += Kmat[(size_t)i * FEATS + j] * v[j];
  red[threadIdx.x] = s;
  __syncthreads();
  for (int off = 128; off > 0; off >>= 1) {
    if (threadIdx.x < off) red[threadIdx.x] += red[threadIdx.x + off];
    __syncthreads();
  }
  if (threadIdx.x == 0) w[i] = red[0];
}

// u[j] = sum_i K[i][j] * w[i]      grid = 8 blocks x 256 (coalesced over j)
__global__ __launch_bounds__(256) void sd_mv_KTw(
    const float* __restrict__ Kmat, const float* __restrict__ w,
    float* __restrict__ u) {
  const int j = blockIdx.x * 256 + threadIdx.x;
  float s = 0.f;
  for (int i = 0; i < D_IN; ++i)
    s += Kmat[(size_t)i * FEATS + j] * w[i];
  u[j] = s;
}

// v = u/||u||;  ||u|| -> sigma^2 at convergence;  invSig = 1/sqrt(||u||)
__global__ __launch_bounds__(256) void sd_normalize(
    const float* __restrict__ u, float* __restrict__ v,
    float* __restrict__ invSig) {
  __shared__ float red[256];
  float s = 0.f;
  for (int j = threadIdx.x; j < FEATS; j += 256) { float x = u[j]; s += x * x; }
  red[threadIdx.x] = s;
  __syncthreads();
  for (int off = 128; off > 0; off >>= 1) {
    if (threadIdx.x < off) red[threadIdx.x] += red[threadIdx.x + off];
    __syncthreads();
  }
  const float nrm = sqrtf(red[0]);             // ||u|| = sigma^2 estimate
  const float inv = 1.0f / nrm;
  for (int j = threadIdx.x; j < FEATS; j += 256) v[j] = u[j] * inv;
  if (threadIdx.x == 0) *invSig = rsqrtf(nrm); // 1/sigma
}

// ------------------------------ main GEMM ----------------------------------
// C = (A @ K) * invSigma ; A f32 row-major [M][K], Bt f16 [N][K], C f32.
__global__ __launch_bounds__(256) void sd_gemm_wmma(
    const float* __restrict__ A, const _Float16* __restrict__ Bt,
    const float* __restrict__ invSigPtr, float* __restrict__ C) {
  __shared__ __align__(16) _Float16 lA[2][BM * LPITCH];   // 2 x 10240 B
  __shared__ __align__(16) _Float16 lB[2][BN * LPITCH];   // 2 x 10240 B

  const int tid    = threadIdx.x;
  const int lane   = tid & 31;
  const int wave   = tid >> 5;           // 0..7
  const int waveM  = wave & 3;           // 4 waves over M (32 rows each)
  const int waveN  = wave >> 2;          // 2 waves over N (64 cols each)
  const int blockM = blockIdx.y * BM;
  const int blockN = blockIdx.x * BN;

  const int l16     = lane & 15;
  const int halfSel = lane >> 4;         // 0: lanes 0-15, 1: lanes 16-31
  const int khalf   = halfSel * 8;       // fragment k offset (halves)

  const float invSigma = *invSigPtr;     // uniform scalar load

  // per-thread staging coordinates
  int rowA[4], c4A[4];                   // A: 1024 float4 chunks (8/row)
#pragma unroll
  for (int i = 0; i < 4; ++i) {
    const int idx = tid + i * 256;
    rowA[i] = idx >> 3;
    c4A[i]  = (idx & 7) * 4;
  }
  int rowB[2], c8B[2];                   // B: 512 v8h chunks (4/row)
#pragma unroll
  for (int i = 0; i < 2; ++i) {
    const int idx = tid + i * 256;
    rowB[i] = idx >> 2;
    c8B[i]  = (idx & 3) * 8;
  }

  float4 aReg[4];
  v8h    bReg[2];

  auto loadGlobal = [&](int k0) {
#pragma unroll
    for (int i = 0; i < 4; ++i)
      aReg[i] = *(const float4*)(A + (size_t)(blockM + rowA[i]) * D_IN + k0 + c4A[i]);
#pragma unroll
    for (int i = 0; i < 2; ++i)
      bReg[i] = *(const v8h*)(Bt + (size_t)(blockN + rowB[i]) * D_IN + k0 + c8B[i]);
  };
  auto storeLds = [&](int b) {
#pragma unroll
    for (int i = 0; i < 4; ++i) {
      v4h h;
      h.x = (_Float16)aReg[i].x; h.y = (_Float16)aReg[i].y;
      h.z = (_Float16)aReg[i].z; h.w = (_Float16)aReg[i].w;
      *(v4h*)(&lA[b][rowA[i] * LPITCH + c4A[i]]) = h;
    }
#pragma unroll
    for (int i = 0; i < 2; ++i)
      *(v8h*)(&lB[b][rowB[i] * LPITCH + c8B[i]]) = bReg[i];
  };

  v8f acc[2][4] = {};

  // prologue: tile 0 -> buffer 0
  loadGlobal(0);
  storeLds(0);
  __syncthreads();

  int buf = 0;
  for (int k0 = 0; k0 < D_IN; k0 += BK) {
    const bool hasNext = (k0 + BK) < D_IN;
    if (hasNext) {
      loadGlobal(k0 + BK);               // overlaps with WMMA below
      if (k0 + 2 * BK < D_IN)            // warm L2 two tiles ahead
        __builtin_prefetch(A + (size_t)(blockM + (tid >> 1)) * D_IN + k0 +
                               2 * BK + (tid & 1) * 16, 0, 1);
    }

    // ---- fragments from current buffer (documented 16-bit VGPR layout) ----
    v16h aF[2], bF[4];
#pragma unroll
    for (int mt = 0; mt < 2; ++mt) {
      const int m = waveM * 32 + mt * 16 + l16;
      const v8h lo = *(const v8h*)(&lA[buf][m * LPITCH + khalf]);
      const v8h hi = *(const v8h*)(&lA[buf][m * LPITCH + 16 + khalf]);
#pragma unroll
      for (int e = 0; e < 8; ++e) { aF[mt][e] = lo[e]; aF[mt][8 + e] = hi[e]; }
    }
#pragma unroll
    for (int nt = 0; nt < 4; ++nt) {
      const int n = waveN * 64 + nt * 16 + l16;
      const v8h lo = *(const v8h*)(&lB[buf][n * LPITCH + khalf]);
      const v8h hi = *(const v8h*)(&lB[buf][n * LPITCH + 16 + khalf]);
#pragma unroll
      for (int e = 0; e < 8; ++e) { bF[nt][e] = lo[e]; bF[nt][8 + e] = hi[e]; }
    }

#pragma unroll
    for (int mt = 0; mt < 2; ++mt)
#pragma unroll
      for (int nt = 0; nt < 4; ++nt)
        acc[mt][nt] = __builtin_amdgcn_wmma_f32_16x16x32_f16(
            false, aF[mt], false, bF[nt], (short)0, acc[mt][nt], false, false);

    if (hasNext) storeLds(buf ^ 1);      // loadcnt wait lands after WMMAs
    __syncthreads();
    buf ^= 1;
  }

  // ---- epilogue: scale by 1/sigma, scatter per documented C layout ----
#pragma unroll
  for (int mt = 0; mt < 2; ++mt) {
#pragma unroll
    for (int nt = 0; nt < 4; ++nt) {
      const int m0 = blockM + waveM * 32 + mt * 16 + halfSel * 8;
      const int n  = blockN + waveN * 64 + nt * 16 + l16;
#pragma unroll
      for (int r = 0; r < 8; ++r)
        C[(size_t)(m0 + r) * FEATS + n] = acc[mt][nt][r] * invSigma;
    }
  }
}

// ------------------------------- launcher ----------------------------------
extern "C" void kernel_launch(void* const* d_in, const int* in_sizes, int n_in,
                              void* d_out, int out_size, void* d_ws,
                              size_t ws_size, hipStream_t stream) {
  const float* A    = (const float*)d_in[0];   // [4,2048,2048] f32
  const float* Kmat = (const float*)d_in[1];   // [2048,2048]  f32
  float*       C    = (float*)d_out;           // [4,2048,2048] f32

  // workspace layout
  char* ws = (char*)d_ws;
  _Float16* Bt = (_Float16*)ws;                         // 8 MiB
  size_t off = (size_t)FEATS * D_IN * sizeof(_Float16);
  float* v  = (float*)(ws + off);  off += D_IN * sizeof(float);
  float* w  = (float*)(ws + off);  off += D_IN * sizeof(float);
  float* u  = (float*)(ws + off);  off += FEATS * sizeof(float);
  float* sg = (float*)(ws + off);

  // 1) transpose+convert kernel matrix once: Bt[n][k] = f16(K[k][n])
  sd_transpose_cvt<<<dim3(FEATS / 32, D_IN / 32), 256, 0, stream>>>(Kmat, Bt);

  // 2) power iteration for top singular value
  sd_init_v<<<FEATS / 256, 256, 0, stream>>>(v);
  for (int it = 0; it < PI_ITERS; ++it) {
    sd_mv_Kv<<<D_IN, 256, 0, stream>>>(Kmat, v, w);
    sd_mv_KTw<<<FEATS / 256, 256, 0, stream>>>(Kmat, w, u);
    sd_normalize<<<1, 256, 0, stream>>>(u, v, sg);
  }

  // 3) WMMA GEMM with 1/sigma folded into epilogue
  sd_gemm_wmma<<<dim3(FEATS / BN, M_TOT / BM), 256, 0, stream>>>(A, Bt, sg, C);
}
